// MoETransformerDecoderLayer_71004399337800
// MI455X (gfx1250) — matrix operand order repaired and checked by reading, not compiled, not run on hardware
//
#include <hip/hip_runtime.h>
#include <hip/hip_bf16.h>
#include <math.h>

typedef _Float16 f16;
typedef __attribute__((ext_vector_type(16))) _Float16 v16h;
typedef __attribute__((ext_vector_type(8)))  float    v8f;

// ---------------------------------------------------------------------------
// WMMA helpers (CDNA5 v_wmma_f32_16x16x32_f16, wave32)
// ---------------------------------------------------------------------------
static __device__ __forceinline__ v8f wmma16(v16h a, v16h b, v8f c) {
  return __builtin_amdgcn_wmma_f32_16x16x32_f16(false, a, false, b, (short)0, c,
                                                false, false);
}

static __device__ __forceinline__ v8f vzero8() {
  v8f z;
#pragma unroll
  for (int i = 0; i < 8; ++i) z[i] = 0.f;
  return z;
}

// Pack two values into one 32-bit word of f16 pairs (for vectorized LDS stores)
union F16x2 { f16 h[2]; unsigned u; };
static __device__ __forceinline__ unsigned pk2f(float a, float b) {
  F16x2 t; t.h[0] = (f16)a; t.h[1] = (f16)b; return t.u;
}
static __device__ __forceinline__ unsigned pk2h(f16 a, f16 b) {
  F16x2 t; t.h[0] = a; t.h[1] = b; return t.u;
}

// A fragment: 16x32 f16 tile, row-major in LDS with row stride ldk.
// Lane half h, row m=lane&15:
//   elems e -> K = 16h + e (matches frag_b)
static __device__ __forceinline__ v16h frag_a(const f16* p0, int ldk, int lane) {
  const int m = lane & 15, h = lane >> 4;
  const f16* p = p0 + m * ldk;
  v16h a;
#pragma unroll
  for (int e = 0; e < 16; ++e) a[e] = p[16 * h + e];
  return a;
}

// B fragment: logical 32x16 B, stored as Bt (N-major [n][k], row stride ldk).
// Lane half h, col n=lane&15: elems e -> K = 16h + e (contiguous 32B load).
static __device__ __forceinline__ v16h frag_b(const f16* p0, int ldk, int lane) {
  const int n = lane & 15, h = lane >> 4;
  const f16* p = p0 + n * ldk + 16 * h;
  v16h b;
#pragma unroll
  for (int e = 0; e < 16; ++e) b[e] = p[e];
  return b;
}

// ---------------------------------------------------------------------------
// Dense GEMM: Out[M,N] = A[M,K] @ W[N,K]^T + bias (+ resid), A/W fp32.
// Block tile 128x64, 8 waves (4x2), wave tile 32x32.
// ---------------------------------------------------------------------------
template <bool OUT_F16, bool RESID>
__global__ __launch_bounds__(256) void gemm_xwT(
    const float* __restrict__ A, int lda,
    const float* __restrict__ W, int ldw,
    const float* __restrict__ bias,
    const float* __restrict__ resid, int ldr,
    void* __restrict__ Out, int ldo, int K) {
  __shared__ __align__(16) f16 As[128][32];
  __shared__ __align__(16) f16 Bs[64][32];
  const int tid = threadIdx.x, lane = tid & 31, wave = tid >> 5;
  const int wm = (wave >> 1) * 32, wn = (wave & 1) * 32;
  const long m0 = (long)blockIdx.y * 128;
  const long n0 = (long)blockIdx.x * 64;

  v8f acc[2][2];
  acc[0][0] = vzero8(); acc[0][1] = vzero8();
  acc[1][0] = vzero8(); acc[1][1] = vzero8();

  const int ar = tid >> 1, ak = (tid & 1) * 16;
  const int br = tid >> 2, bk = (tid & 3) * 8;

  for (int k0 = 0; k0 < K; k0 += 32) {
    const float* as = A + (m0 + ar) * lda + k0 + ak;
    {
      uint4 w0, w1;
      w0.x = pk2f(as[0], as[1]);   w0.y = pk2f(as[2], as[3]);
      w0.z = pk2f(as[4], as[5]);   w0.w = pk2f(as[6], as[7]);
      w1.x = pk2f(as[8], as[9]);   w1.y = pk2f(as[10], as[11]);
      w1.z = pk2f(as[12], as[13]); w1.w = pk2f(as[14], as[15]);
      uint4* dst = (uint4*)&As[ar][ak];
      dst[0] = w0; dst[1] = w1;
    }
    const float* bsrc = W + (n0 + br) * ldw + k0 + bk;
    {
      uint4 wb;
      wb.x = pk2f(bsrc[0], bsrc[1]); wb.y = pk2f(bsrc[2], bsrc[3]);
      wb.z = pk2f(bsrc[4], bsrc[5]); wb.w = pk2f(bsrc[6], bsrc[7]);
      *(uint4*)&Bs[br][bk] = wb;
    }
    if (k0 + 32 < K) {  // lowers to global_prefetch_b8
      __builtin_prefetch(as + 32);
      __builtin_prefetch(bsrc + 32);
    }
    __syncthreads();
    v16h a0 = frag_a(&As[wm][0], 32, lane);
    v16h a1 = frag_a(&As[wm + 16][0], 32, lane);
    v16h b0 = frag_b(&Bs[wn][0], 32, lane);
    v16h b1 = frag_b(&Bs[wn + 16][0], 32, lane);
    acc[0][0] = wmma16(a0, b0, acc[0][0]);
    acc[0][1] = wmma16(a0, b1, acc[0][1]);
    acc[1][0] = wmma16(a1, b0, acc[1][0]);
    acc[1][1] = wmma16(a1, b1, acc[1][1]);
    __syncthreads();
  }
  const int nl = lane & 15, hf = lane >> 4;
#pragma unroll
  for (int ti = 0; ti < 2; ++ti)
#pragma unroll
    for (int tj = 0; tj < 2; ++tj)
#pragma unroll
      for (int r = 0; r < 8; ++r) {
        long m = m0 + wm + ti * 16 + r + 8 * hf;
        long n = n0 + wn + tj * 16 + nl;
        float v = acc[ti][tj][r] + bias[n];
        if (RESID) v += resid[m * (long)ldr + n];
        if (OUT_F16) ((f16*)Out)[m * (long)ldo + n] = (f16)v;
        else         ((float*)Out)[m * (long)ldo + n] = v;
      }
}

// ---------------------------------------------------------------------------
// Flash attention: block = (128 q rows, one head, one batch); 8 waves x 16 q.
// Q/K/V are f16, head-strided inside packed projection buffers.
// Softmax scale is folded into the Q staging.
// ---------------------------------------------------------------------------
__global__ __launch_bounds__(256) void attn_wmma(
    const f16* __restrict__ Q, int ldq,
    const f16* __restrict__ Kx, int ldk,
    const f16* __restrict__ V, int ldv,
    float* __restrict__ O, int Lq, int Lk, float scale) {
  __shared__ __align__(16) f16 Qs[128][64];
  __shared__ __align__(16) f16 Ks[32][64];
  __shared__ __align__(16) f16 Vt[64][32];
  __shared__ __align__(16) f16 Ps[8][16][32];

  const int tid = threadIdx.x, lane = tid & 31, wave = tid >> 5;
  const int b = blockIdx.z, h = blockIdx.y;
  const long q0 = (long)blockIdx.x * 128;
  const f16* Qb = Q + (long)b * Lq * ldq + h * 64;
  const f16* Kb = Kx + (long)b * Lk * ldk + h * 64;
  const f16* Vb = V + (long)b * Lk * ldv + h * 64;

  {  // stage Q (pre-scaled): 32 f16 per thread -> packed b32 stores
    int r = tid >> 1, c = (tid & 1) * 32;
    const f16* src = Qb + (q0 + r) * (long)ldq + c;
    unsigned* qd = (unsigned*)&Qs[r][c];
#pragma unroll
    for (int i = 0; i < 16; ++i)
      qd[i] = pk2f(scale * (float)src[2 * i], scale * (float)src[2 * i + 1]);
  }
  __syncthreads();

  v16h qf0 = frag_a(&Qs[wave * 16][0], 64, lane);
  v16h qf1 = frag_a(&Qs[wave * 16][32], 64, lane);

  float mi[8], li[8];
#pragma unroll
  for (int r = 0; r < 8; ++r) { mi[r] = -3.0e38f; li[r] = 0.f; }
  v8f acc[4];
#pragma unroll
  for (int t = 0; t < 4; ++t) acc[t] = vzero8();

  const int sr = tid >> 3, sc = (tid & 7) * 8;    // K staging: 16B per thread
  const int sr2 = (tid >> 4) * 2;                 // V staging: 2 kpos rows
  const int sc4 = (tid & 15) * 4;                 // 4 hd cols
  const int nl = lane & 15, hf = lane >> 4;

  for (int kc = 0; kc < Lk; kc += 32) {
    const f16* ks = Kb + (long)(kc + sr) * ldk + sc;
    *(uint4*)&Ks[sr][sc] = *(const uint4*)ks;     // raw 128-bit copy
    const f16* v0 = Vb + (long)(kc + sr2) * ldv + sc4;
    const f16* v1 = v0 + ldv;
#pragma unroll
    for (int i = 0; i < 4; ++i)                   // transposed, packed b32
      *(unsigned*)&Vt[sc4 + i][sr2] = pk2h(v0[i], v1[i]);
    if (kc + 32 < Lk) {
      __builtin_prefetch(ks + 32 * (long)ldk);
      __builtin_prefetch(v0 + 32 * (long)ldv);
    }
    __syncthreads();

    v8f s0 = vzero8(), s1 = vzero8();
    {
      v16h kb0 = frag_b(&Ks[0][0], 64, lane);
      v16h kb1 = frag_b(&Ks[0][32], 64, lane);
      s0 = wmma16(qf0, kb0, s0);
      s0 = wmma16(qf1, kb1, s0);
      v16h kb2 = frag_b(&Ks[16][0], 64, lane);
      v16h kb3 = frag_b(&Ks[16][32], 64, lane);
      s1 = wmma16(qf0, kb2, s1);
      s1 = wmma16(qf1, kb3, s1);
    }
    float p0[8], p1[8], al[8];
#pragma unroll
    for (int r = 0; r < 8; ++r) {
      float sa = s0[r], sb = s1[r];               // already scaled via Q
      float cm = fmaxf(sa, sb);
#pragma unroll
      for (int off = 1; off < 16; off <<= 1) cm = fmaxf(cm, __shfl_xor(cm, off, 32));
      float mn = fmaxf(mi[r], cm);
      al[r] = __expf(mi[r] - mn);
      p0[r] = __expf(sa - mn);
      p1[r] = __expf(sb - mn);
      float rs = p0[r] + p1[r];
#pragma unroll
      for (int off = 1; off < 16; off <<= 1) rs += __shfl_xor(rs, off, 32);
      li[r] = li[r] * al[r] + rs;
      mi[r] = mn;
    }
#pragma unroll
    for (int t = 0; t < 4; ++t)
#pragma unroll
      for (int r = 0; r < 8; ++r) acc[t][r] = acc[t][r] * al[r];
#pragma unroll
    for (int r = 0; r < 8; ++r) {
      Ps[wave][r + 8 * hf][nl] = (f16)p0[r];
      Ps[wave][r + 8 * hf][16 + nl] = (f16)p1[r];
    }
    __syncthreads();
    v16h pf = frag_a(&Ps[wave][0][0], 32, lane);
#pragma unroll
    for (int t = 0; t < 4; ++t) {
      v16h vb = frag_b(&Vt[t * 16][0], 32, lane);
      acc[t] = wmma16(pf, vb, acc[t]);
    }
    __syncthreads();
  }
#pragma unroll
  for (int t = 0; t < 4; ++t)
#pragma unroll
    for (int r = 0; r < 8; ++r) {
      long q = q0 + wave * 16 + r + 8 * hf;
      O[((long)b * Lq + q) * 512 + h * 64 + t * 16 + nl] = acc[t][r] / li[r];
    }
}

// ---------------------------------------------------------------------------
// LayerNorm over D=512, one wave per row.
// ---------------------------------------------------------------------------
__global__ __launch_bounds__(256) void layernorm512(
    const float* __restrict__ X, const float* __restrict__ G,
    const float* __restrict__ Bb, float* __restrict__ Y) {
  const int tid = threadIdx.x, lane = tid & 31, wave = tid >> 5;
  const long row = (long)blockIdx.x * 8 + wave;
  const float* x = X + row * 512;
  float v[16], s = 0.f, s2 = 0.f;
#pragma unroll
  for (int i = 0; i < 16; ++i) {
    float t = x[lane + i * 32];
    v[i] = t; s += t; s2 += t * t;
  }
#pragma unroll
  for (int off = 1; off < 32; off <<= 1) {
    s += __shfl_xor(s, off, 32);
    s2 += __shfl_xor(s2, off, 32);
  }
  float mu = s * (1.f / 512.f);
  float var = s2 * (1.f / 512.f) - mu * mu;
  float rstd = rsqrtf(var + 1e-5f);
  float* y = Y + row * 512;
#pragma unroll
  for (int i = 0; i < 16; ++i) {
    int c = lane + i * 32;
    y[c] = (v[i] - mu) * rstd * G[c] + Bb[c];
  }
}

// ---------------------------------------------------------------------------
// MoE routing: top-1 argmax of gate logits (softmax preserves argmax).
// ---------------------------------------------------------------------------
__global__ void moe_reset(int* counts) {
  if (threadIdx.x < 4) counts[threadIdx.x] = 0;
}

__global__ __launch_bounds__(256) void gate_route(
    const float* __restrict__ X, const float* __restrict__ GW,
    const float* __restrict__ GB, int* __restrict__ counts,
    int* __restrict__ elist, int Mtok) {
  const int tid = threadIdx.x, lane = tid & 31, wave = tid >> 5;
  const long t = (long)blockIdx.x * 8 + wave;
  const float* x = X + t * 512;
  float a0 = 0, a1 = 0, a2 = 0, a3 = 0;
  for (int i = lane; i < 512; i += 32) {
    float xv = x[i];
    a0 += xv * GW[i];
    a1 += xv * GW[512 + i];
    a2 += xv * GW[1024 + i];
    a3 += xv * GW[1536 + i];
  }
#pragma unroll
  for (int off = 1; off < 32; off <<= 1) {
    a0 += __shfl_xor(a0, off, 32);
    a1 += __shfl_xor(a1, off, 32);
    a2 += __shfl_xor(a2, off, 32);
    a3 += __shfl_xor(a3, off, 32);
  }
  if (lane == 0) {
    float l[4] = {a0 + GB[0], a1 + GB[1], a2 + GB[2], a3 + GB[3]};
    int best = 0; float bv = l[0];
#pragma unroll
    for (int e = 1; e < 4; ++e)
      if (l[e] > bv) { bv = l[e]; best = e; }
    int pos = atomicAdd(&counts[best], 1);
    elist[best * Mtok + pos] = (int)t;
  }
}

__global__ void moe_scan(const int* counts, int* offs) {
  if (threadIdx.x == 0) {
    int o = 0;
    for (int e = 0; e < 4; ++e) { offs[e] = o; o += counts[e]; }
    offs[4] = o;
  }
}

// ---------------------------------------------------------------------------
// MoE FFN1: H[slot, 2048] = gelu(gather(X) @ W1[e] + b1[e]) ; W1 is K-major.
// ---------------------------------------------------------------------------
__global__ __launch_bounds__(256) void moe_ffn1(
    const float* __restrict__ X, const float* __restrict__ W1,
    const float* __restrict__ B1, f16* __restrict__ H,
    const int* __restrict__ counts, const int* __restrict__ offs,
    const int* __restrict__ elist, int Mtok) {
  __shared__ __align__(16) f16 As[128][32];
  __shared__ __align__(16) f16 Bs[64][32];
  const int e = blockIdx.z;
  const int cnt = counts[e];
  const int m0 = blockIdx.y * 128;
  if (m0 >= cnt) return;
  const int n0 = blockIdx.x * 64;
  const float* W = W1 + (long)e * 512 * 2048;
  const int tid = threadIdx.x, lane = tid & 31, wave = tid >> 5;
  const int wm = (wave >> 1) * 32, wn = (wave & 1) * 32;

  v8f acc[2][2];
  acc[0][0] = vzero8(); acc[0][1] = vzero8();
  acc[1][0] = vzero8(); acc[1][1] = vzero8();

  const int ar = tid >> 1, ak = (tid & 1) * 16;
  const int bk2 = (tid >> 4) * 2;   // pair of K rows
  const int bn4 = (tid & 15) * 4;   // 4 N cols
  int tokA = (m0 + ar < cnt) ? elist[e * Mtok + m0 + ar] : -1;

  for (int k0 = 0; k0 < 512; k0 += 32) {
    uint4* adst = (uint4*)&As[ar][ak];
    if (tokA >= 0) {
      const float* src = X + (long)tokA * 512 + k0 + ak;
      uint4 w0, w1;
      w0.x = pk2f(src[0], src[1]);   w0.y = pk2f(src[2], src[3]);
      w0.z = pk2f(src[4], src[5]);   w0.w = pk2f(src[6], src[7]);
      w1.x = pk2f(src[8], src[9]);   w1.y = pk2f(src[10], src[11]);
      w1.z = pk2f(src[12], src[13]); w1.w = pk2f(src[14], src[15]);
      adst[0] = w0; adst[1] = w1;
    } else {
      uint4 z; z.x = z.y = z.z = z.w = 0u;
      adst[0] = z; adst[1] = z;
    }
    const float* br0 = W + (long)(k0 + bk2) * 2048 + n0 + bn4;
    const float* br1 = br0 + 2048;
#pragma unroll
    for (int i = 0; i < 4; ++i)
      *(unsigned*)&Bs[bn4 + i][bk2] = pk2f(br0[i], br1[i]);
    if (k0 + 32 < 512) __builtin_prefetch(br0 + 32 * 2048);
    __syncthreads();
    v16h a0 = frag_a(&As[wm][0], 32, lane);
    v16h a1 = frag_a(&As[wm + 16][0], 32, lane);
    v16h b0 = frag_b(&Bs[wn][0], 32, lane);
    v16h b1 = frag_b(&Bs[wn + 16][0], 32, lane);
    acc[0][0] = wmma16(a0, b0, acc[0][0]);
    acc[0][1] = wmma16(a0, b1, acc[0][1]);
    acc[1][0] = wmma16(a1, b0, acc[1][0]);
    acc[1][1] = wmma16(a1, b1, acc[1][1]);
    __syncthreads();
  }
  const int nl = lane & 15, hf = lane >> 4;
  const float* bias = B1 + (long)e * 2048;
#pragma unroll
  for (int ti = 0; ti < 2; ++ti)
#pragma unroll
    for (int tj = 0; tj < 2; ++tj)
#pragma unroll
      for (int r = 0; r < 8; ++r) {
        int ml = wm + ti * 16 + r + 8 * hf;
        if (m0 + ml < cnt) {
          int n = n0 + wn + tj * 16 + nl;
          float v = acc[ti][tj][r] + bias[n];
          v = 0.5f * v * (1.f + erff(v * 0.70710678f));  // exact GELU
          H[(long)(offs[e] + m0 + ml) * 2048 + n] = (f16)v;
        }
      }
}

// ---------------------------------------------------------------------------
// MoE FFN2: Y[token, 512] = H[slot] @ W2[e] + b2[e] + Xres[token] (scatter).
// ---------------------------------------------------------------------------
__global__ __launch_bounds__(256) void moe_ffn2(
    const f16* __restrict__ H, const float* __restrict__ W2,
    const float* __restrict__ B2, const float* __restrict__ Xres,
    float* __restrict__ Y, const int* __restrict__ counts,
    const int* __restrict__ offs, const int* __restrict__ elist, int Mtok) {
  __shared__ __align__(16) f16 As[128][32];
  __shared__ __align__(16) f16 Bs[64][32];
  const int e = blockIdx.z;
  const int cnt = counts[e];
  const int m0 = blockIdx.y * 128;
  if (m0 >= cnt) return;
  const int n0 = blockIdx.x * 64;
  const float* W = W2 + (long)e * 2048 * 512;
  const int tid = threadIdx.x, lane = tid & 31, wave = tid >> 5;
  const int wm = (wave >> 1) * 32, wn = (wave & 1) * 32;

  v8f acc[2][2];
  acc[0][0] = vzero8(); acc[0][1] = vzero8();
  acc[1][0] = vzero8(); acc[1][1] = vzero8();

  const int ar = tid >> 1, ak = (tid & 1) * 16;
  const int bk2 = (tid >> 4) * 2;
  const int bn4 = (tid & 15) * 4;
  const bool aval = (m0 + ar < cnt);
  const long slotA = (long)offs[e] + m0 + ar;

  for (int k0 = 0; k0 < 2048; k0 += 32) {
    uint4* adst = (uint4*)&As[ar][ak];
    if (aval) {
      const uint4* src = (const uint4*)(H + slotA * 2048 + k0 + ak);
      adst[0] = src[0]; adst[1] = src[1];
      if (k0 + 32 < 2048) __builtin_prefetch(src + 4);
    } else {
      uint4 z; z.x = z.y = z.z = z.w = 0u;
      adst[0] = z; adst[1] = z;
    }
    const float* br0 = W + (long)(k0 + bk2) * 512 + n0 + bn4;
    const float* br1 = br0 + 512;
#pragma unroll
    for (int i = 0; i < 4; ++i)
      *(unsigned*)&Bs[bn4 + i][bk2] = pk2f(br0[i], br1[i]);
    if (k0 + 32 < 2048) __builtin_prefetch(br0 + 32 * 512);
    __syncthreads();
    v16h a0 = frag_a(&As[wm][0], 32, lane);
    v16h a1 = frag_a(&As[wm + 16][0], 32, lane);
    v16h b0 = frag_b(&Bs[wn][0], 32, lane);
    v16h b1 = frag_b(&Bs[wn + 16][0], 32, lane);
    acc[0][0] = wmma16(a0, b0, acc[0][0]);
    acc[0][1] = wmma16(a0, b1, acc[0][1]);
    acc[1][0] = wmma16(a1, b0, acc[1][0]);
    acc[1][1] = wmma16(a1, b1, acc[1][1]);
    __syncthreads();
  }
  const int nl = lane & 15, hf = lane >> 4;
  const float* bias = B2 + (long)e * 512;
#pragma unroll
  for (int ti = 0; ti < 2; ++ti)
#pragma unroll
    for (int tj = 0; tj < 2; ++tj)
#pragma unroll
      for (int r = 0; r < 8; ++r) {
        int ml = wm + ti * 16 + r + 8 * hf;
        if (m0 + ml < cnt) {
          int tok = elist[e * Mtok + m0 + ml];
          int n = n0 + wn + tj * 16 + nl;
          float v = acc[ti][tj][r] + bias[n] + Xres[(long)tok * 512 + n];
          Y[(long)tok * 512 + n] = v;
        }
      }
}

// ---------------------------------------------------------------------------
// Host launcher
// ---------------------------------------------------------------------------
extern "C" void kernel_launch(void* const* d_in, const int* in_sizes, int n_in,
                              void* d_out, int out_size, void* d_ws,
                              size_t ws_size, hipStream_t stream) {
  (void)in_sizes; (void)n_in; (void)out_size; (void)ws_size;
  const float* x        = (const float*)d_in[0];
  const float* memory   = (const float*)d_in[1];
  const float* sa_in_w  = (const float*)d_in[2];
  const float* sa_in_b  = (const float*)d_in[3];
  const float* sa_out_w = (const float*)d_in[4];
  const float* sa_out_b = (const float*)d_in[5];
  const float* ca_in_w  = (const float*)d_in[6];
  const float* ca_in_b  = (const float*)d_in[7];
  const float* ca_out_w = (const float*)d_in[8];
  const float* ca_out_b = (const float*)d_in[9];
  const float* gate_w   = (const float*)d_in[10];
  const float* gate_b   = (const float*)d_in[11];
  const float* w1       = (const float*)d_in[12];
  const float* b1       = (const float*)d_in[13];
  const float* w2       = (const float*)d_in[14];
  const float* b2       = (const float*)d_in[15];
  const float* ln1_g    = (const float*)d_in[16];
  const float* ln1_b    = (const float*)d_in[17];
  const float* ln2_g    = (const float*)d_in[18];
  const float* ln2_b    = (const float*)d_in[19];
  const float* ln3_g    = (const float*)d_in[20];
  const float* ln3_b    = (const float*)d_in[21];

  const int LT = 2048, Bq = 4;
  const int Mtok = Bq * LT;  // 8192 (== B*LM as well)

  char* ws = (char*)d_ws;
  size_t off = 0;
  auto take = [&](size_t bytes) -> char* {
    char* p = ws + off;
    off = (off + bytes + 255) & ~(size_t)255;
    return p;
  };
  f16*   qkv    = (f16*)take((size_t)Mtok * 1536 * sizeof(f16));
  float* attn_o = (float*)take((size_t)Mtok * 512 * sizeof(float));
  float* sum    = (float*)take((size_t)Mtok * 512 * sizeof(float));
  float* x1     = (float*)take((size_t)Mtok * 512 * sizeof(float));
  float* x2     = (float*)take((size_t)Mtok * 512 * sizeof(float));
  f16*   hbuf   = (f16*)take((size_t)Mtok * 2048 * sizeof(f16));
  int*   elist  = (int*)take((size_t)4 * Mtok * sizeof(int));
  int*   counts = (int*)take(64);
  int*   offs   = (int*)take(64);
  f16* q2  = qkv;                       // reuse QKV buffer after self-attn
  f16* kv2 = qkv + (size_t)Mtok * 512;

  dim3 blk(256);

  // 1) self-attn QKV projection -> f16 [8192,1536]
  gemm_xwT<true, false><<<dim3(1536 / 64, Mtok / 128), blk, 0, stream>>>(
      x, 512, sa_in_w, 512, sa_in_b, nullptr, 0, qkv, 1536, 512);
  // 2) self attention
  attn_wmma<<<dim3(LT / 128, 8, Bq), blk, 0, stream>>>(
      qkv, 1536, qkv + 512, 1536, qkv + 1024, 1536, attn_o, LT, LT, 0.125f);
  // 3) self out-proj + residual(x) -> sum
  gemm_xwT<false, true><<<dim3(512 / 64, Mtok / 128), blk, 0, stream>>>(
      attn_o, 512, sa_out_w, 512, sa_out_b, x, 512, sum, 512, 512);
  // 4) LN1 -> x1
  layernorm512<<<dim3(Mtok / 8), blk, 0, stream>>>(sum, ln1_g, ln1_b, x1);
  // 5) cross Q projection (x1, first 512 rows of ca_in_w) -> q2 f16
  gemm_xwT<true, false><<<dim3(512 / 64, Mtok / 128), blk, 0, stream>>>(
      x1, 512, ca_in_w, 512, ca_in_b, nullptr, 0, q2, 512, 512);
  // 6) cross K/V projection (memory, rows 512..1535) -> kv2 f16 [8192,1024]
  gemm_xwT<true, false><<<dim3(1024 / 64, Mtok / 128), blk, 0, stream>>>(
      memory, 512, ca_in_w + 512 * 512, 512, ca_in_b + 512, nullptr, 0, kv2,
      1024, 512);
  // 7) cross attention
  attn_wmma<<<dim3(LT / 128, 8, Bq), blk, 0, stream>>>(
      q2, 512, kv2, 1024, kv2 + 512, 1024, attn_o, LT, LT, 0.125f);
  // 8) cross out-proj + residual(x1) -> sum
  gemm_xwT<false, true><<<dim3(512 / 64, Mtok / 128), blk, 0, stream>>>(
      attn_o, 512, ca_out_w, 512, ca_out_b, x1, 512, sum, 512, 512);
  // 9) LN2 -> x2
  layernorm512<<<dim3(Mtok / 8), blk, 0, stream>>>(sum, ln2_g, ln2_b, x2);
  // 10) top-1 routing
  moe_reset<<<1, 32, 0, stream>>>(counts);
  gate_route<<<dim3(Mtok / 8), blk, 0, stream>>>(x2, gate_w, gate_b, counts,
                                                 elist, Mtok);
  moe_scan<<<1, 32, 0, stream>>>(counts, offs);
  // 11) expert FFN1 (grouped, gathered) -> hbuf f16
  moe_ffn1<<<dim3(2048 / 64, Mtok / 128, 4), blk, 0, stream>>>(
      x2, w1, b1, hbuf, counts, offs, elist, Mtok);
  // 12) expert FFN2 (+residual x2, scatter) -> sum
  moe_ffn2<<<dim3(512 / 64, Mtok / 128, 4), blk, 0, stream>>>(
      hbuf, w2, b2, x2, sum, counts, offs, elist, Mtok);
  // 13) LN3 -> out
  layernorm512<<<dim3(Mtok / 8), blk, 0, stream>>>(sum, ln3_g, ln3_b,
                                                   (float*)d_out);
}
